// MultiHeadAttn_78357383348549
// MI455X (gfx1250) — matrix operand order repaired and checked by reading, not compiled, not run on hardware
//
#include <hip/hip_runtime.h>
#include <hip/hip_bf16.h>

typedef __attribute__((ext_vector_type(16))) _Float16 v16h;
typedef __attribute__((ext_vector_type(8)))  _Float16 v8h;
typedef __attribute__((ext_vector_type(8)))  float    v8f;

#define WMMA_F16(A, B, C) \
  __builtin_amdgcn_wmma_f32_16x16x32_f16(false, (A), false, (B), (short)0, (C), false, false)

static constexpr int B_ = 4, L_ = 2048, D_ = 1024, H_ = 16, DH_ = 64;

__device__ inline v8f zero_v8f() {
  v8f r;
#pragma unroll
  for (int i = 0; i < 8; i++) r[i] = 0.0f;
  return r;
}

// A-operand (16x32 f16, M x K): lane m = lane&15, g = lane>>4.
// halves 0..7  -> K = 16g .. 16g+7   (contiguous)
// halves 8..15 -> K = 16g+8 .. 16g+15 (contiguous)
__device__ inline v16h make_a16(const _Float16* lo_p, const _Float16* hi_p) {
  v8h lo = *(const v8h*)lo_p;
  v8h hi = *(const v8h*)hi_p;
  v16h r;
#pragma unroll
  for (int i = 0; i < 8; i++) { r[i] = lo[i]; r[i + 8] = hi[i]; }
  return r;
}

// Same A pattern but converting from f32 source (activations)
__device__ inline v16h cvt_a16_f32(const float* lo_p, const float* hi_p) {
  v8f lo = *(const v8f*)lo_p;
  v8f hi = *(const v8f*)hi_p;
  v16h r;
#pragma unroll
  for (int i = 0; i < 8; i++) { r[i] = (_Float16)lo[i]; r[i + 8] = (_Float16)hi[i]; }
  return r;
}

// ---------------------------------------------------------------------------
// Kernel 0: one-shot weight conversion f32 -> f16 (wq|wk|wv concatenated).
// ---------------------------------------------------------------------------
__global__ __launch_bounds__(256)
void wcvt_kernel(const float* __restrict__ wq, const float* __restrict__ wk,
                 const float* __restrict__ wv, _Float16* __restrict__ wf) {
  const size_t t = (size_t)blockIdx.x * blockDim.x + threadIdx.x;
  const size_t per = (size_t)D_ * D_;           // 1M elements per matrix
  const size_t idx = t * 8;                      // element index in [0, 3M)
  const int mat = (int)(idx / per);
  const size_t off = idx % per;
  const float* src = (mat == 0) ? wq : (mat == 1) ? wk : wv;
  v8f x = *(const v8f*)(src + off);
  v8h y;
#pragma unroll
  for (int i = 0; i < 8; i++) y[i] = (_Float16)x[i];
  *(v8h*)(wf + idx) = y;
}

// ---------------------------------------------------------------------------
// Kernel 1: fused QKV projection.  Y[n,e] = sum_d X[n,d] * W[e,d]  (x @ W^T)
// Each wave computes a 16-row x 128-col tile with 8 WMMA accumulators.
// Software-pipelined: all 8 B-operands staged first (one load clause),
// then 8 back-to-back WMMAs (no per-WMMA loadcnt wait).
// Q,K stored f16 as [b,h,l,64]; V stored transposed f16 as [b,h,64,l].
// 1/sqrt(dh)=0.125 folded into Q.
// ---------------------------------------------------------------------------
__global__ __launch_bounds__(256)
void mha_proj_kernel(const float* __restrict__ q_in, const float* __restrict__ k_in,
                     const float* __restrict__ v_in, const _Float16* __restrict__ wf,
                     _Float16* __restrict__ qh, _Float16* __restrict__ kh,
                     _Float16* __restrict__ vt) {
  const int wave = (blockIdx.x * blockDim.x + threadIdx.x) >> 5;
  const int lane = threadIdx.x & 31;
  const int g = lane >> 4, n = lane & 15;

  // 512 row-tiles (of B*L=8192 rows), 8 col-groups of 128 -> 4096 waves/matrix
  const int mat = wave >> 12;
  const int rem = wave & 4095;
  const int rt  = rem >> 3;
  const int cg  = rem & 7;

  const float* X = (mat == 0) ? q_in : (mat == 1) ? k_in : v_in;
  const _Float16* W = wf + (size_t)mat * D_ * D_;
  const float scale = (mat == 0) ? 0.125f : 1.0f;

  v8f acc[8];
#pragma unroll
  for (int e = 0; e < 8; e++) acc[e] = zero_v8f();

  const int row0 = rt * 16;
  const float* xrow = X + (size_t)(row0 + n) * D_;          // A lane: row m = n
  const _Float16* wbase = W + (size_t)(cg * 128 + n) * D_;  // B lane: col n

  for (int d0 = 0; d0 < D_; d0 += 32) {
    // Stage all B-operands first: 16 independent b128 loads in flight.
    v16h b[8];
#pragma unroll
    for (int e = 0; e < 8; e++)
      b[e] = *(const v16h*)(wbase + (size_t)e * 16 * D_ + d0 + 16 * g);
    v16h a = cvt_a16_f32(xrow + d0 + 16 * g, xrow + d0 + 16 * g + 8);
    // Back-to-back WMMAs (independent accumulators, shared A).
#pragma unroll
    for (int e = 0; e < 8; e++)
      acc[e] = WMMA_F16(a, b[e], acc[e]);
  }

  const int bb = row0 >> 11;      // / 2048
  const int l0 = row0 & 2047;
#pragma unroll
  for (int e = 0; e < 8; e++) {
    const int ecol = cg * 128 + e * 16 + n;
    const int h  = ecol >> 6;
    const int dd = ecol & 63;
#pragma unroll
    for (int r = 0; r < 8; r++) {
      const int m = r + 8 * g;
      const int l = l0 + m;
      const _Float16 val = (_Float16)(acc[e][r] * scale);
      if (mat == 2) {
        vt[(((size_t)bb * H_ + h) * DH_ + dd) * L_ + l] = val;
      } else {
        _Float16* dst = (mat == 0) ? qh : kh;
        dst[(((size_t)bb * H_ + h) * L_ + l) * DH_ + dd] = val;
      }
    }
  }
}

// ---------------------------------------------------------------------------
// Kernel 2: fused attention. One wave owns a 16-query tile of one (b,h).
// Pass 1: stream 128 K-tiles through WMMA, online row max/sum.
// Pass 2: recompute scores, write normalized attn (the HBM-bound output),
//         stage P->LDS->A-layout, accumulate out += P @ V with WMMA.
// ---------------------------------------------------------------------------
__device__ inline v8f score_tile(const _Float16* kbase, int j, int n, int g,
                                 v16h aq0, v16h aq1) {
  const _Float16* krow = kbase + (size_t)(j * 16 + n) * DH_;
  v16h b0 = *(const v16h*)(krow + 16 * g);
  v16h b1 = *(const v16h*)(krow + 32 + 16 * g);
  v8f s = zero_v8f();
  s = WMMA_F16(aq0, b0, s);
  s = WMMA_F16(aq1, b1, s);
  return s;
}

__global__ __launch_bounds__(256)
void mha_attn_kernel(const _Float16* __restrict__ qh, const _Float16* __restrict__ kh,
                     const _Float16* __restrict__ vt, float* __restrict__ out,
                     float* __restrict__ attn) {
  __shared__ __align__(64) _Float16 lds_p[8][16][32];

  const int wblk = threadIdx.x >> 5;
  const int wave = blockIdx.x * 8 + wblk;
  const int lane = threadIdx.x & 31;
  const int g = lane >> 4, n = lane & 15;

  const int bh = wave >> 7;   // b*16 + h
  const int qt = wave & 127;  // q tile (16 rows each)

  const _Float16* qbase = qh + ((size_t)bh * L_ + qt * 16) * DH_;
  const _Float16* kbase = kh + (size_t)bh * L_ * DH_;
  const _Float16* vbase = vt + (size_t)bh * DH_ * L_;
  float* attn_base = attn + ((size_t)bh * L_ + qt * 16) * L_;

  // Q A-operands (dh=64 -> two 16x32 chunks), scale already folded in
  const _Float16* qrow = qbase + (size_t)n * DH_;
  const v16h aq0 = make_a16(qrow + 16 * g,      qrow + 16 * g + 8);
  const v16h aq1 = make_a16(qrow + 32 + 16 * g, qrow + 40 + 16 * g);

  float mrow[8], lrow[8];
#pragma unroll
  for (int r = 0; r < 8; r++) { mrow[r] = -1e30f; lrow[r] = 0.0f; }

  // ---- Pass 1: online softmax statistics ----
  for (int j = 0; j < L_ / 16; j++) {
    v8f s = score_tile(kbase, j, n, g, aq0, aq1);
#pragma unroll
    for (int r = 0; r < 8; r++) {
      float tm = s[r];
      tm = fmaxf(tm, __shfl_xor(tm, 1));
      tm = fmaxf(tm, __shfl_xor(tm, 2));
      tm = fmaxf(tm, __shfl_xor(tm, 4));
      tm = fmaxf(tm, __shfl_xor(tm, 8));
      const float nm = fmaxf(mrow[r], tm);
      float ts = __expf(s[r] - nm);
      ts += __shfl_xor(ts, 1);
      ts += __shfl_xor(ts, 2);
      ts += __shfl_xor(ts, 4);
      ts += __shfl_xor(ts, 8);
      lrow[r] = lrow[r] * __expf(mrow[r] - nm) + ts;
      mrow[r] = nm;
    }
  }
  float inv_l[8];
#pragma unroll
  for (int r = 0; r < 8; r++) inv_l[r] = 1.0f / lrow[r];

  v8f oacc[4];
#pragma unroll
  for (int c = 0; c < 4; c++) oacc[c] = zero_v8f();

  // ---- Pass 2: write attn, accumulate out = P @ V (k in chunks of 32) ----
  for (int j2 = 0; j2 < L_ / 32; j2++) {
#pragma unroll
    for (int jj = 0; jj < 2; jj++) {
      const int j = j2 * 2 + jj;
      v8f s = score_tile(kbase, j, n, g, aq0, aq1);
#pragma unroll
      for (int r = 0; r < 8; r++) {
        const float p = __expf(s[r] - mrow[r]) * inv_l[r];
        attn_base[(size_t)(r + 8 * g) * L_ + j * 16 + n] = p;   // required output
        lds_p[wblk][r + 8 * g][jj * 16 + n] = (_Float16)p;      // stage for PV
      }
    }
    // Stage all V B-operands (independent of the LDS round-trip latency).
    v16h bv[4];
#pragma unroll
    for (int c = 0; c < 4; c++)
      bv[c] = *(const v16h*)(vbase + (size_t)(c * 16 + n) * L_ + j2 * 32 + 16 * g);
    asm volatile("s_wait_dscnt 0" ::: "memory");  // cross-lane LDS RAW (in-wave)
    const _Float16* prow = &lds_p[wblk][n][0];
    v16h ap = make_a16(prow + 16 * g, prow + 16 * g + 8);
#pragma unroll
    for (int c = 0; c < 4; c++)
      oacc[c] = WMMA_F16(ap, bv[c], oacc[c]);
  }

  // ---- Epilogue: out[b, l, h*64 + d] ----
  const int bb = bh >> 4;
  const int h  = bh & 15;
#pragma unroll
  for (int c = 0; c < 4; c++) {
#pragma unroll
    for (int r = 0; r < 8; r++) {
      const int m = r + 8 * g;
      out[((size_t)bb * L_ + qt * 16 + m) * D_ + h * DH_ + c * 16 + n] = oacc[c][r];
    }
  }
}

extern "C" void kernel_launch(void* const* d_in, const int* in_sizes, int n_in,
                              void* d_out, int out_size, void* d_ws, size_t ws_size,
                              hipStream_t stream) {
  const float* q  = (const float*)d_in[0];
  const float* k  = (const float*)d_in[1];
  const float* v  = (const float*)d_in[2];
  const float* wq = (const float*)d_in[3];
  const float* wk = (const float*)d_in[4];
  const float* wv = (const float*)d_in[5];

  const size_t per_tensor = (size_t)B_ * H_ * L_ * DH_;  // 8.4M halves = 16 MB
  _Float16* qh = (_Float16*)d_ws;
  _Float16* kh = qh + per_tensor;
  _Float16* vt = kh + per_tensor;
  _Float16* wf = vt + per_tensor;                        // 3*1M halves = 6 MB

  float* out  = (float*)d_out;                     // [B, L, D]
  float* attn = out + (size_t)B_ * L_ * D_;        // [B, H, L, L]

  // Weight f32 -> f16 (3M elements, 8 per thread -> 393216 threads)
  wcvt_kernel<<<1536, 256, 0, stream>>>(wq, wk, wv, wf);
  // Projection: 3 matrices * 512 row-tiles * 8 col-groups = 12288 waves
  mha_proj_kernel<<<12288 / 8, 256, 0, stream>>>(q, k, v, wf, qh, kh, vt);
  // Attention: B*H*(L/16) = 8192 waves
  mha_attn_kernel<<<8192 / 8, 256, 0, stream>>>(qh, kh, vt, out, attn);
}
